// GPTSmall_46986942218512
// MI455X (gfx1250) — matrix operand order repaired and checked by reading, not compiled, not run on hardware
//
#include <hip/hip_runtime.h>
#include <math.h>

// ---------------------------------------------------------------------------
// GPT-small forward (B=2, T=1024, D=512, H=8, L=2, V=50257) for gfx1250.
// All GEMMs run on v_wmma_f32_16x16x32_bf16 (bf16 in, f32 accumulate): the
// tied LM head (105 GFLOP of ~140 GFLOP total) makes this compute-bound;
// ~110MB of weights vs 23.3TB/s HBM => memory ~5us, so the matrix pipes are
// the bottleneck. Staging uses gfx1250 async memory->LDS copies (ASYNCcnt)
// with double-buffered LDS (K-slice of 64 => 8 WMMAs per barrier).
// ---------------------------------------------------------------------------

typedef __attribute__((ext_vector_type(16))) __bf16 bf16x16;
typedef __attribute__((ext_vector_type(8)))  float  floatx8;

union FragAB { bf16x16 v; unsigned int u[8]; };
union FragC  { floatx8 v; float f[8]; };

__device__ __forceinline__ unsigned short f32_to_bf16_rne(float x) {
  unsigned int u = __builtin_bit_cast(unsigned int, x);
  unsigned int r = u + 0x7FFFu + ((u >> 16) & 1u);   // round-to-nearest-even
  return (unsigned short)(r >> 16);
}

// LDS byte offset of a shared-memory pointer (generic -> AS3 -> int).
__device__ __forceinline__ unsigned lds_off(const void* p) {
  return (unsigned)(unsigned long long)(__attribute__((address_space(3))) const void*)p;
}

// ------------------------------ GEMM (WMMA) --------------------------------
// C[M,N] = A[M,K](bf16, row-major) * W[N,K](bf16, row-major, torch [out,in])^T
//          (+ bias[N] if non-null), C is f32.
// Block: 256 threads = 8 waves; tile 64(M) x 128(N); K staged in slices of 64
// (two 16x16x32 WMMA sub-steps per slice => 8 WMMAs per wave per barrier).
// Double-buffered LDS; full tiles staged with global_load_async_to_lds_b128.
#define GEMM_BM 64
#define GEMM_BN 128
#define GEMM_BK 64

__global__ __launch_bounds__(256) void gpt_gemm_bf16_wmma(
    const unsigned short* __restrict__ A,
    const unsigned short* __restrict__ W,
    const float* __restrict__ bias,
    float* __restrict__ C,
    int Mrows, int Ncols, int Kdim)
{
  __shared__ __align__(16) unsigned short sA[2][GEMM_BM][GEMM_BK]; // 2x8 KB
  __shared__ __align__(16) unsigned short sB[2][GEMM_BN][GEMM_BK]; // 2x16 KB

  const int tid  = threadIdx.x;
  const int lane = tid & 31;
  const int wid  = tid >> 5;     // 0..7
  const int wr   = wid >> 2;     // 0..1 : 32-row slab
  const int wc   = wid & 3;      // 0..3 : 32-col slab
  const int blockM = blockIdx.y * GEMM_BM;
  const int blockN = blockIdx.x * GEMM_BN;
  const bool mFull = (blockM + GEMM_BM) <= Mrows;   // uniform per block
  const bool nFull = (blockN + GEMM_BN) <= Ncols;   // uniform per block

  // staging: 16B (8 bf16) per chunk; A slice = 512 chunks, B slice = 1024
  auto stageA = [&](int k0, int buf) {
    #pragma unroll
    for (int it = 0; it < 2; ++it) {
      int chunk = tid + it * 256;            // 0..511
      int r = chunk >> 3;                    // A row 0..63
      int c = (chunk & 7) * 8;               // k-col {0,8,...,56}
      const unsigned short* src = A + (size_t)(blockM + r) * Kdim + k0 + c;
      if (mFull) {
        asm volatile("global_load_async_to_lds_b128 %0, %1, off"
                     :: "v"(lds_off(&sA[buf][r][c])), "v"(src) : "memory");
      } else {
        uint4 val; val.x = val.y = val.z = val.w = 0u;
        if (blockM + r < Mrows) val = *(const uint4*)src;
        *(uint4*)&sA[buf][r][c] = val;
      }
    }
  };
  auto stageB = [&](int k0, int buf) {
    #pragma unroll
    for (int it = 0; it < 4; ++it) {
      int chunk = tid + it * 256;            // 0..1023
      int r = chunk >> 3;                    // B row(n) 0..127
      int c = (chunk & 7) * 8;
      const unsigned short* src = W + (size_t)(blockN + r) * Kdim + k0 + c;
      if (nFull) {
        asm volatile("global_load_async_to_lds_b128 %0, %1, off"
                     :: "v"(lds_off(&sB[buf][r][c])), "v"(src) : "memory");
      } else {
        uint4 val; val.x = val.y = val.z = val.w = 0u;
        if (blockN + r < Ncols) val = *(const uint4*)src;
        *(uint4*)&sB[buf][r][c] = val;
      }
    }
  };
  auto wait_stage = [&]() {
    if (mFull || nFull)
      asm volatile("s_wait_asynccnt 0x0" ::: "memory");
  };

  FragC acc[2][2];
  #pragma unroll
  for (int i = 0; i < 2; ++i)
    #pragma unroll
    for (int j = 0; j < 2; ++j)
      #pragma unroll
      for (int r = 0; r < 8; ++r) acc[i][j].f[r] = 0.0f;

  // prologue: stage K-slice 0 into buffer 0
  stageA(0, 0);
  stageB(0, 0);
  wait_stage();
  __syncthreads();

  const int lrow  = lane & 15;
  const int khalf = lane >> 4;          // 0 or 1
  const int akb   = khalf * 8;          // A: K interleave base (ISA 7.12.2)
  const int bkb   = khalf * 16;         // unused: B is read with akb, same k order as A

  int p = 0;
  for (int k0 = 0; k0 < Kdim; k0 += GEMM_BK) {
    const int kn = k0 + GEMM_BK;
    const bool hasNext = kn < Kdim;
    if (hasNext) {                       // prefetch next slice into other buf
      stageA(kn, p ^ 1);
      stageB(kn, p ^ 1);
    }

    // --- two 16x16x32 K sub-steps on the staged 64-wide slice
    #pragma unroll
    for (int ss = 0; ss < 2; ++ss) {
      const int kb = ss * 32;
      FragAB afrag[2], bfrag[2];
      #pragma unroll
      for (int t2 = 0; t2 < 2; ++t2) {
        const unsigned short* pa = &sA[p][wr * 32 + t2 * 16 + lrow][kb];
        #pragma unroll
        for (int i = 0; i < 8; ++i) {
          int k = (i < 4) ? (akb + 2 * i) : (16 + akb + 2 * (i - 4));
          afrag[t2].u[i] = *(const unsigned int*)(pa + k);
        }
        const unsigned short* pb = &sB[p][wc * 32 + t2 * 16 + lrow][kb];
        #pragma unroll
        for (int i = 0; i < 8; ++i) {
          int k = (i < 4) ? (akb + 2 * i) : (16 + akb + 2 * (i - 4));
          bfrag[t2].u[i] = *(const unsigned int*)(pb + k);
        }
      }
      #pragma unroll
      for (int i = 0; i < 2; ++i)
        #pragma unroll
        for (int j = 0; j < 2; ++j)
          acc[i][j].v = __builtin_amdgcn_wmma_f32_16x16x32_bf16(
              false, afrag[i].v, false, bfrag[j].v,
              (short)0, acc[i][j].v, false, false);
    }

    if (hasNext) wait_stage();
    __syncthreads();
    p ^= 1;
  }

  // --- store (C/D layout: vgpr r -> M = r + 8*khalf, N = lane&15)
  const int mtop = khalf * 8;
  if (mFull && nFull) {
    #pragma unroll
    for (int i = 0; i < 2; ++i) {
      #pragma unroll
      for (int j = 0; j < 2; ++j) {
        int n = blockN + wc * 32 + j * 16 + lrow;
        float bi = (bias != nullptr) ? bias[n] : 0.0f;
        #pragma unroll
        for (int r = 0; r < 8; ++r) {
          int m = blockM + wr * 32 + i * 16 + mtop + r;
          C[(size_t)m * Ncols + n] = acc[i][j].f[r] + bi;
        }
      }
    }
  } else {
    #pragma unroll
    for (int i = 0; i < 2; ++i) {
      #pragma unroll
      for (int j = 0; j < 2; ++j) {
        int n = blockN + wc * 32 + j * 16 + lrow;
        float bi = 0.0f;
        if (bias != nullptr && n < Ncols) bi = bias[n];
        #pragma unroll
        for (int r = 0; r < 8; ++r) {
          int m = blockM + wr * 32 + i * 16 + mtop + r;
          if (m < Mrows && n < Ncols)
            C[(size_t)m * Ncols + n] = acc[i][j].f[r] + bi;
        }
      }
    }
  }
}

// --------------------------- block reductions ------------------------------
__device__ __forceinline__ float blk_sum(float v, volatile float* red) {
  __syncthreads();
  #pragma unroll
  for (int off = 16; off; off >>= 1) v += __shfl_xor(v, off, 32);
  int lane = threadIdx.x & 31, wid = threadIdx.x >> 5;
  if (lane == 0) red[wid] = v;
  __syncthreads();
  if (threadIdx.x == 0) {
    float s = 0.0f;
    int nw = blockDim.x >> 5;
    for (int w = 0; w < nw; ++w) s += red[w];
    red[0] = s;
  }
  __syncthreads();
  return red[0];
}

__device__ __forceinline__ float blk_max(float v, volatile float* red) {
  __syncthreads();
  #pragma unroll
  for (int off = 16; off; off >>= 1) v = fmaxf(v, __shfl_xor(v, off, 32));
  int lane = threadIdx.x & 31, wid = threadIdx.x >> 5;
  if (lane == 0) red[wid] = v;
  __syncthreads();
  if (threadIdx.x == 0) {
    float s = -3.4e38f;
    int nw = blockDim.x >> 5;
    for (int w = 0; w < nw; ++w) s = fmaxf(s, red[w]);
    red[0] = s;
  }
  __syncthreads();
  return red[0];
}

// ------------------------------ elementwise --------------------------------
__global__ void gpt_embed(const int* __restrict__ idx,
                          const float* __restrict__ tok,
                          const float* __restrict__ pos,
                          float* __restrict__ x, int T, int D)
{
  int i = blockIdx.x;            // over B*T
  int t = i % T;
  int tk = idx[i];
  for (int d = threadIdx.x; d < D; d += blockDim.x)
    x[(size_t)i * D + d] = tok[(size_t)tk * D + d] + pos[(size_t)t * D + d];
}

__global__ __launch_bounds__(256) void gpt_layernorm_bf16(
    const float* __restrict__ x, const float* __restrict__ w,
    const float* __restrict__ b, unsigned short* __restrict__ out, int D)
{
  __shared__ float red[8];
  int row = blockIdx.x;
  const float* xr = x + (size_t)row * D;
  float s = 0.0f, s2 = 0.0f;
  for (int d = threadIdx.x; d < D; d += blockDim.x) {
    float v = xr[d]; s += v; s2 += v * v;
  }
  float S  = blk_sum(s, red);
  float S2 = blk_sum(s2, red);
  float mu = S / (float)D;
  float var = S2 / (float)D - mu * mu;
  float rstd = rsqrtf(var + 1e-5f);
  unsigned short* orow = out + (size_t)row * D;
  for (int d = threadIdx.x; d < D; d += blockDim.x)
    orow[d] = f32_to_bf16_rne((xr[d] - mu) * rstd * w[d] + b[d]);
}

__global__ void gpt_f32_to_bf16(const float* __restrict__ in,
                                unsigned short* __restrict__ out, size_t n)
{
  size_t i = (size_t)blockIdx.x * blockDim.x + threadIdx.x;
  size_t stride = (size_t)gridDim.x * blockDim.x;
  for (; i < n; i += stride) out[i] = f32_to_bf16_rne(in[i]);
}

__global__ void gpt_gelu_to_bf16(const float* __restrict__ in,
                                 unsigned short* __restrict__ out, size_t n)
{
  size_t i = (size_t)blockIdx.x * blockDim.x + threadIdx.x;
  size_t stride = (size_t)gridDim.x * blockDim.x;
  for (; i < n; i += stride) {
    float v = in[i];
    float g = 0.5f * v * (1.0f + erff(v * 0.70710678118654752f)); // exact gelu
    out[i] = f32_to_bf16_rne(g);
  }
}

__global__ void gpt_residual_add(float* __restrict__ x,
                                 const float* __restrict__ t, size_t n)
{
  size_t i = (size_t)blockIdx.x * blockDim.x + threadIdx.x;
  size_t stride = (size_t)gridDim.x * blockDim.x;
  for (; i < n; i += stride) x[i] += t[i];
}

// ----------------------------- attention -----------------------------------
// One wave32 per (b,h,t) query row; causal softmax over j<=t with LDS score
// row + wave shuffles; each lane accumulates 2 of the 64 head dims.
__global__ __launch_bounds__(128) void gpt_attention(
    const float* __restrict__ qkv, float* __restrict__ y,
    int T, int H, int HD, int D)
{
  __shared__ float sc[4][1024];
  int wid  = threadIdx.x >> 5;
  int lane = threadIdx.x & 31;
  int gw = blockIdx.x * 4 + wid;
  int t = gw % T;
  int h = (gw / T) % H;
  int b = gw / (T * H);
  float* s = sc[wid];

  const size_t row3D = (size_t)3 * D;
  const float* qrow = qkv + (size_t)(b * T + t) * row3D + h * HD;
  float scale = rsqrtf((float)HD);

  float mx = -3.4e38f;
  for (int j = lane; j <= t; j += 32) {
    const float* krow = qkv + (size_t)(b * T + j) * row3D + D + h * HD;
    float d = 0.0f;
    for (int e = 0; e < HD; ++e) d += qrow[e] * krow[e];
    d *= scale;
    s[j] = d;
    mx = fmaxf(mx, d);
  }
  #pragma unroll
  for (int off = 16; off; off >>= 1) mx = fmaxf(mx, __shfl_xor(mx, off, 32));

  float sum = 0.0f;
  for (int j = lane; j <= t; j += 32) {
    float e = expf(s[j] - mx);
    s[j] = e;
    sum += e;
  }
  #pragma unroll
  for (int off = 16; off; off >>= 1) sum += __shfl_xor(sum, off, 32);
  float inv = 1.0f / sum;

  __syncthreads();  // make every lane's s[j] visible

  float a0 = 0.0f, a1 = 0.0f;
  int e0 = lane * 2;
  for (int j = 0; j <= t; ++j) {
    float p = s[j];
    const float* vrow = qkv + (size_t)(b * T + j) * row3D + 2 * D + h * HD;
    a0 += p * vrow[e0];
    a1 += p * vrow[e0 + 1];
  }
  float* yr = y + (size_t)(b * T + t) * D + h * HD;
  yr[e0]     = a0 * inv;
  yr[e0 + 1] = a1 * inv;
}

// ------------------------------- loss --------------------------------------
__global__ void gpt_zero_loss(float* p) { *p = 0.0f; }

__global__ __launch_bounds__(256) void gpt_ce_loss(
    const float* __restrict__ logits, const int* __restrict__ targets,
    float* __restrict__ loss, int V, float invBT)
{
  __shared__ float red[8];
  int row = blockIdx.x;
  const float* lr = logits + (size_t)row * V;
  float mx = -3.4e38f;
  for (int i = threadIdx.x; i < V; i += blockDim.x) mx = fmaxf(mx, lr[i]);
  mx = blk_max(mx, red);
  float s = 0.0f;
  for (int i = threadIdx.x; i < V; i += blockDim.x) s += expf(lr[i] - mx);
  s = blk_sum(s, red);
  if (threadIdx.x == 0) {
    float lse = mx + logf(s);
    atomicAdd(loss, (lse - lr[targets[row]]) * invBT);
  }
}

// ------------------------------- driver ------------------------------------
extern "C" void kernel_launch(void* const* d_in, const int* in_sizes, int n_in,
                              void* d_out, int out_size, void* d_ws, size_t ws_size,
                              hipStream_t stream) {
  constexpr int V = 50257, D = 512, H = 8, L = 2, T = 1024, B = 2, HD = 64;
  constexpr int BT = B * T;
  (void)in_sizes; (void)n_in; (void)out_size; (void)ws_size;

  const int*   idx     = (const int*)  d_in[0];
  const int*   targets = (const int*)  d_in[1];
  const float* tok_emb = (const float*)d_in[2];
  const float* pos_emb = (const float*)d_in[3];
  const float* ln1_w   = (const float*)d_in[4];
  const float* ln1_b   = (const float*)d_in[5];
  const float* attn_w  = (const float*)d_in[6];
  const float* attn_b  = (const float*)d_in[7];
  const float* proj_w  = (const float*)d_in[8];
  const float* proj_b  = (const float*)d_in[9];
  const float* ln2_w   = (const float*)d_in[10];
  const float* ln2_b   = (const float*)d_in[11];
  const float* fc_w    = (const float*)d_in[12];
  const float* fc_b    = (const float*)d_in[13];
  const float* fc2_w   = (const float*)d_in[14];
  const float* fc2_b   = (const float*)d_in[15];
  const float* lnf_w   = (const float*)d_in[16];
  const float* lnf_b   = (const float*)d_in[17];

  char* ws = (char*)d_ws;
  size_t off = 0;
  auto alloc = [&](size_t bytes) -> void* {
    void* p = ws + off;
    off = (off + bytes + 255) & ~(size_t)255;
    return p;
  };

  float* xbuf   = (float*)alloc(sizeof(float) * BT * D);
  float* qkvbuf = (float*)alloc(sizeof(float) * BT * 3 * D);
  float* ybuf   = (float*)alloc(sizeof(float) * BT * D);
  float* tmpbuf = (float*)alloc(sizeof(float) * BT * D);
  float* fcf    = (float*)alloc(sizeof(float) * BT * 4 * D);
  unsigned short* hbf   = (unsigned short*)alloc(sizeof(short) * BT * D);
  unsigned short* ybf   = (unsigned short*)alloc(sizeof(short) * BT * D);
  unsigned short* fcbf  = (unsigned short*)alloc(sizeof(short) * BT * 4 * D);
  unsigned short* wqkv  = (unsigned short*)alloc(sizeof(short) * (size_t)L * 3 * D * D);
  unsigned short* wproj = (unsigned short*)alloc(sizeof(short) * (size_t)L * D * D);
  unsigned short* wfc   = (unsigned short*)alloc(sizeof(short) * (size_t)L * 4 * D * D);
  unsigned short* wfc2  = (unsigned short*)alloc(sizeof(short) * (size_t)L * D * 4 * D);
  unsigned short* wtok  = (unsigned short*)alloc(sizeof(short) * (size_t)V * D);

  auto conv = [&](const float* in, unsigned short* out, size_t n) {
    int blocks = (int)((n + 255) / 256);
    if (blocks > 8192) blocks = 8192;
    gpt_f32_to_bf16<<<blocks, 256, 0, stream>>>(in, out, n);
  };
  auto gemm = [&](const unsigned short* A, const unsigned short* Wm,
                  const float* bias, float* C, int M, int N, int K) {
    dim3 grid((N + GEMM_BN - 1) / GEMM_BN, (M + GEMM_BM - 1) / GEMM_BM);
    gpt_gemm_bf16_wmma<<<grid, 256, 0, stream>>>(A, Wm, bias, C, M, N, K);
  };
  auto eadd = [&](float* x, const float* t, size_t n) {
    int blocks = (int)((n + 255) / 256);
    if (blocks > 8192) blocks = 8192;
    gpt_residual_add<<<blocks, 256, 0, stream>>>(x, t, n);
  };

  // weight precision conversion (deterministic every launch)
  conv(attn_w, wqkv,  (size_t)L * 3 * D * D);
  conv(proj_w, wproj, (size_t)L * D * D);
  conv(fc_w,   wfc,   (size_t)L * 4 * D * D);
  conv(fc2_w,  wfc2,  (size_t)L * D * 4 * D);
  conv(tok_emb, wtok, (size_t)V * D);

  // embedding
  gpt_embed<<<BT, 256, 0, stream>>>(idx, tok_emb, pos_emb, xbuf, T, D);

  for (int l = 0; l < L; ++l) {
    // attention block
    gpt_layernorm_bf16<<<BT, 256, 0, stream>>>(xbuf, ln1_w + l * D, ln1_b + l * D, hbf, D);
    gemm(hbf, wqkv + (size_t)l * 3 * D * D, attn_b + l * 3 * D, qkvbuf, BT, 3 * D, D);
    gpt_attention<<<(B * H * T) / 4, 128, 0, stream>>>(qkvbuf, ybuf, T, H, HD, D);
    conv(ybuf, ybf, (size_t)BT * D);
    gemm(ybf, wproj + (size_t)l * D * D, proj_b + l * D, tmpbuf, BT, D, D);
    eadd(xbuf, tmpbuf, (size_t)BT * D);
    // MLP block
    gpt_layernorm_bf16<<<BT, 256, 0, stream>>>(xbuf, ln2_w + l * D, ln2_b + l * D, hbf, D);
    gemm(hbf, wfc + (size_t)l * 4 * D * D, fc_b + l * 4 * D, fcf, BT, 4 * D, D);
    {
      size_t n = (size_t)BT * 4 * D;
      int blocks = (int)((n + 255) / 256);
      if (blocks > 8192) blocks = 8192;
      gpt_gelu_to_bf16<<<blocks, 256, 0, stream>>>(fcf, fcbf, n);
    }
    gemm(fcbf, wfc2 + (size_t)l * D * 4 * D, fc2_b + l * D, tmpbuf, BT, D, 4 * D);
    eadd(xbuf, tmpbuf, (size_t)BT * D);
  }

  // final LN + tied LM head (the 105-GFLOP GEMM) straight into d_out
  gpt_layernorm_bf16<<<BT, 256, 0, stream>>>(xbuf, lnf_w, lnf_b, hbf, D);
  float* logits = (float*)d_out;
  gemm(hbf, wtok, nullptr, logits, BT, V, D);

  // cross-entropy
  float* loss_ptr = logits + (size_t)BT * V;
  gpt_zero_loss<<<1, 1, 0, stream>>>(loss_ptr);
  gpt_ce_loss<<<BT, 256, 0, stream>>>(logits, targets, loss_ptr, V, 1.0f / (float)BT);
}
